// RoutingCaps_78245714198907
// MI455X (gfx1250) — matrix operand order repaired and checked by reading, not compiled, not run on hardware
//
#include <hip/hip_runtime.h>
#include <hip/hip_fp16.h>

// ---------------------------------------------------------------------------
// RoutingCaps fused pipeline for MI455X (gfx1250, wave32, WMMA + TDM)
//
//   Kcvt : f32 -> f16 copies of x1..x3 and W1..W3 (tiny)
//   Kweff: collapse 65-tap depthwise-grouped conv + row-sum into
//          Weff[o][i*16+m] = sum_{k=m+17..m+32} w[o,i,k]
//   KA   : batched GEMM (M=16 batch rows, N=256 class cols, K=128) via
//          v_wmma_f32_16x16x32_f16; W/x tiles staged with tensor_load_to_lds
//          (TDM, double-buffered, per-wave -> zero barriers in GEMM loop);
//          fused focus epilogue; usum + focused u1/u2/u3 (f16) -> workspace
//   KB   : per-item linear attention + conv-collapse + squash
// ---------------------------------------------------------------------------

typedef _Float16 h16 __attribute__((ext_vector_type(16)));
typedef _Float16 h8  __attribute__((ext_vector_type(8)));
typedef float    f8  __attribute__((ext_vector_type(8)));
typedef unsigned int u32x4 __attribute__((ext_vector_type(4)));
typedef int          i32x8 __attribute__((ext_vector_type(8)));
typedef int          i32x4 __attribute__((ext_vector_type(4)));

#define BATCH   16384
#define NK      128         // GEMM K (N0*D0)
#define NCOL    2304        // CLS*16*16
#define CLSN    9
#define NROW    144         // N1*D1
#define EPSF32  1.1920928955078125e-07f

// ---------------------------------------------------------------------------
// TDM: issue a 2D tile load (f16 elements) global -> LDS.
// D# layout per CDNA5 ISA ch.8: group0 = {count|flags, lds_addr, gaddr_lo,
// gaddr_hi|type}, group1 = {mask/data_size, tensor dims, tile dims, strides}.
// Tensor dims set huge so no OOB clipping occurs (tile is always in-bounds).
// This toolchain exposes the 6-arg builtin; groups 2/3 + trailing group are
// zero (2D tile, no iteration/gather).
// ---------------------------------------------------------------------------
__device__ __forceinline__ void tdm_load_tile_f16(unsigned lds_addr, const void* gaddr,
                                                  unsigned tile_d0, unsigned tile_d1,
                                                  unsigned long long stride0_elems) {
    unsigned long long ga = (unsigned long long)(size_t)gaddr;
    const unsigned td0 = 1u << 20, td1 = 1u << 20;   // "tensor" dims: huge
    u32x4 g0;
    g0[0] = 1u;                                      // count=1, user descriptor
    g0[1] = lds_addr;                                // LDS byte address
    g0[2] = (unsigned)(ga & 0xFFFFFFFFu);            // global addr [31:0]
    g0[3] = (unsigned)((ga >> 32) & 0x01FFFFFFu)     // global addr [56:32]
          | (2u << 30);                              // type=2 ("image")
    i32x8 g1;
    g1[0] = (int)(1u << 16);                                      // data_size=1 (2B), mask=0
    g1[1] = (int)((td0 & 0xFFFFu) << 16);                         // tensor_dim0 lo
    g1[2] = (int)(((td0 >> 16) & 0xFFFFu) | ((td1 & 0xFFFFu) << 16));
    g1[3] = (int)(((td1 >> 16) & 0xFFFFu) | ((tile_d0 & 0xFFFFu) << 16));
    g1[4] = (int)(tile_d1 & 0xFFFFu);                             // tile_dim2 = 0
    g1[5] = (int)(unsigned)(stride0_elems & 0xFFFFFFFFu);         // dim0 stride lo
    g1[6] = (int)(unsigned)((stride0_elems >> 32) & 0xFFFFu);     // stride hi | dim1_stride=0
    g1[7] = 0;
    i32x4 z4 = {0, 0, 0, 0};
    i32x8 z8 = {0, 0, 0, 0, 0, 0, 0, 0};
    __builtin_amdgcn_tensor_load_to_lds(g0, g1, z4, z4, z8, 0);
}

// ---------------------------------------------------------------------------
// Kernel: convert inputs and weights to f16
// ---------------------------------------------------------------------------
__global__ void rc_cvt_kernel(const float* __restrict__ x1, const float* __restrict__ x2,
                              const float* __restrict__ x3, const float* __restrict__ W1,
                              const float* __restrict__ W2, const float* __restrict__ W3,
                              _Float16* __restrict__ xf, _Float16* __restrict__ wf) {
    const long long NX = 3LL * BATCH * NK;
    const long long NW = 3LL * NCOL * NK;
    long long tid = (long long)blockIdx.x * blockDim.x + threadIdx.x;
    if (tid < NX) {
        const long long per = (long long)BATCH * NK;
        int m = (int)(tid / per); long long r = tid % per;
        const float* s = (m == 0) ? x1 : ((m == 1) ? x2 : x3);
        xf[tid] = (_Float16)s[r];
    } else if (tid < NX + NW) {
        long long idx = tid - NX;
        const long long per = (long long)NCOL * NK;
        int m = (int)(idx / per); long long r = idx % per;
        const float* s = (m == 0) ? W1 : ((m == 1) ? W2 : W3);
        wf[idx] = (_Float16)s[r];
    }
}

// ---------------------------------------------------------------------------
// Kernel: effective conv weights (conv + sum over h collapsed)
// ---------------------------------------------------------------------------
__global__ void rc_weff_kernel(const float* __restrict__ dwc_w, float* __restrict__ weff) {
    int id = blockIdx.x * blockDim.x + threadIdx.x;
    if (id >= NROW * 16) return;
    int o = id >> 4, i = id & 15;
    const float* wr = dwc_w + (size_t)(o * 16 + i) * 65;
    float s = 0.f;
    #pragma unroll
    for (int k = 17; k <= 32; ++k) s += wr[k];
    weff[o * 256 + i * 16 + 0] = s;
    #pragma unroll
    for (int m = 1; m < 16; ++m) {
        s += wr[m + 32] - wr[m + 16];
        weff[o * 256 + i * 16 + m] = s;
    }
}

// ---------------------------------------------------------------------------
// Kernel A: WMMA GEMM + focus.  grid = (B/16, 9), block = 96 (3 waves).
// Wave w owns matrix w: x tile (TDM) + double-buffered W chunks (TDM).
// LDS: xs [3][16][128] f16 (12KB) | per-wave [2][256][32] f16 (3*32KB);
//      accumulator dump [16][256] f32 aliases the wave's own buffer pair.
// ---------------------------------------------------------------------------
__global__ void __launch_bounds__(96)
rc_gemm_focus_kernel(const _Float16* __restrict__ xf,   // [3][B][128]
                     const _Float16* __restrict__ wf,   // [3][2304][128]
                     _Float16* __restrict__ usumg,      // [B][2304]
                     _Float16* __restrict__ ufoc)       // [B][3][144][16]
{
    __shared__ __align__(128) unsigned char smem[12288 + 3 * 32768];
    _Float16* xs = (_Float16*)smem;                       // [3][16][128]

    const int tid  = threadIdx.x;
    const int wv   = tid >> 5;        // matrix id 0..2
    const int lane = tid & 31;
    const int b0   = blockIdx.x * 16;
    const int cls  = blockIdx.y;

    _Float16* wbuf0 = (_Float16*)(smem + 12288 + wv * 32768);
    _Float16* wbuf1 = wbuf0 + 8192;                       // [256][32] each

    // ---- TDM: stage this wave's x tile (contiguous 2048 halves) ----
    tdm_load_tile_f16((unsigned)(size_t)(xs + wv * 2048),
                      xf + ((size_t)wv * BATCH + b0) * NK, 2048, 1, 2048);
    // ---- TDM: first W chunk ([256 cols][32 k], row stride 128) ----
    const _Float16* wbase = wf + ((size_t)wv * NCOL + (size_t)cls * 256) * NK;
    tdm_load_tile_f16((unsigned)(size_t)wbuf0, wbase, 32, 256, NK);

    f8 acc[16] = {};
    const int mrow  = lane & 15;
    const int abase = (lane & 16) ? 16 : 0;   // A-frag K base (contiguous k = 16h + s, same as B)
    const int kkb   = (lane & 16) ? 16 : 0;   // B-frag K base

    for (int kc = 0; kc < 4; ++kc) {          // K chunks of 32
        _Float16* wcur = (kc & 1) ? wbuf1 : wbuf0;
        if (kc < 3) {
            // prefetch next chunk into the other buffer, then wait for current
            _Float16* wnxt = (kc & 1) ? wbuf0 : wbuf1;
            asm volatile("s_wait_dscnt 0x0" ::: "memory");   // prior reads of wnxt drained
            tdm_load_tile_f16((unsigned)(size_t)wnxt, wbase + (size_t)(kc + 1) * 32,
                              32, 256, NK);
            __builtin_amdgcn_s_wait_tensorcnt(1);            // in-order => current done
        } else {
            __builtin_amdgcn_s_wait_tensorcnt(0);
        }
        asm volatile("" ::: "memory");        // keep LDS reads below the wait

        // A fragment: two contiguous 8-half segments per lane
        const _Float16* arow = xs + (wv * 16 + mrow) * NK + kc * 32 + abase;
        h8 alo = *reinterpret_cast<const h8*>(arow);
        h8 ahi = *reinterpret_cast<const h8*>(arow + 8);
        h16 afrag;
        #pragma unroll
        for (int e = 0; e < 8; ++e) { afrag[e] = alo[e]; afrag[e + 8] = ahi[e]; }

        #pragma unroll
        for (int nt = 0; nt < 16; ++nt) {
            const _Float16* bp = wcur + (nt * 16 + (lane & 15)) * 32 + kkb;
            h16 bfrag = *reinterpret_cast<const h16*>(bp);
            acc[nt] = __builtin_amdgcn_wmma_f32_16x16x32_f16(
                false, afrag, false, bfrag, (short)0, acc[nt], false, false);
        }
    }

    // ---- dump accumulators into own region (aliases own wbuf0; last reads
    //      were from wbuf1, and DS ops from one wave are in-order) ----
    {
        float* ub = (float*)(smem + 12288 + wv * 32768);   // [16][256] f32
        const int moff = (lane & 16) ? 8 : 0;
        const int ncol = lane & 15;
        #pragma unroll
        for (int nt = 0; nt < 16; ++nt)
            #pragma unroll
            for (int r = 0; r < 8; ++r)
                ub[(r + moff) * 256 + nt * 16 + ncol] = acc[nt][r];
    }
    __syncthreads();   // epilogue reads all three waves' u-blocks

    const float* ub0 = (const float*)(smem + 12288);
    const float* ub1 = (const float*)(smem + 12288 + 32768);
    const float* ub2 = (const float*)(smem + 12288 + 65536);

    // ---- usum = u11+u22+u33, store f16 ----
    for (int e = tid; e < 4096; e += 96) {
        float s = ub0[e] + ub1[e] + ub2[e];
        int row = e >> 8, col = e & 255;
        usumg[(size_t)(b0 + row) * NCOL + cls * 256 + col] = (_Float16)s;
    }

    // ---- focus: per (matrix m, batch row, q): vector over p of u[.., p*16+q] ----
    for (int v = tid; v < 768; v += 96) {
        int m = v >> 8, rq = v & 255, row = rq >> 4, q = rq & 15;
        const float* ub = (m == 0) ? ub0 : ((m == 1) ? ub1 : ub2);
        float a[16]; float n2 = 0.f, n6 = 0.f;
        #pragma unroll
        for (int p = 0; p < 16; ++p) {
            float u  = ub[row * 256 + p * 16 + q];
            float g  = 0.5f * u * (1.0f + erff(u * 0.70710678118654752f)); // exact GELU
            float av = fabsf(g + 1e-6f) * 0.2f;
            a[p] = av;
            n2 += av * av;
            float a3 = av * av * av;
            n6 += a3 * a3;
        }
        float scale = sqrtf(n2) / sqrtf(n6);       // ||a|| / ||a^3||
        size_t base = (((size_t)(b0 + row) * 3 + m) * NCOL) + (size_t)(cls * 16 + q) * 16;
        #pragma unroll
        for (int p = 0; p < 16; ++p)
            ufoc[base + p] = (_Float16)(a[p] * a[p] * a[p] * scale);
    }
}

// ---------------------------------------------------------------------------
// Kernel B: per-item attention + conv-collapse + squash. block = 160 threads.
// ---------------------------------------------------------------------------
__global__ void __launch_bounds__(160)
rc_attn_kernel(const _Float16* __restrict__ ufoc,   // [B][3][144][16]
               const _Float16* __restrict__ usumg,  // [B][2304]
               const float* __restrict__ weff,      // [144][256]
               const float* __restrict__ dwc_b,     // [144]
               float* __restrict__ out)             // [B][9][16]
{
    __shared__ float u1s[NCOL], u2s[NCOL], u3s[NCOL];
    __shared__ float xb[NROW * 16];
    __shared__ float usm[NCOL];
    __shared__ float rx[NROW], tb[NROW], cs[16], kv[256], sb[NROW];

    const int b = blockIdx.x, t = threadIdx.x;

    const _Float16* uf = ufoc + (size_t)b * 3 * NCOL;
    for (int e = t; e < 3 * NCOL; e += 160) {
        float v = (float)uf[e];
        int m = e / NCOL, r = e % NCOL;
        float* dst = (m == 0) ? u1s : ((m == 1) ? u2s : u3s);
        dst[r] = v;
    }
    for (int e = t; e < NCOL; e += 160)
        usm[e] = (float)usumg[(size_t)b * NCOL + e];
    __syncthreads();

    if (t < 16) {                         // column sums of u1 (over 144 rows)
        float s = 0.f;
        for (int j = 0; j < NROW; ++j) s += u1s[j * 16 + t];
        cs[t] = s;
    }
    for (int e = t; e < 256; e += 160) {  // kv = u2^T u3
        int c = e >> 4, d = e & 15; float s = 0.f;
        for (int j = 0; j < NROW; ++j) s += u2s[j * 16 + c] * u3s[j * 16 + d];
        kv[e] = s;
    }
    __syncthreads();

    if (t < NROW) {                       // x = diag(z) * u1 * kv ; row sums
        float ur[16]; float zden = EPSF32;
        #pragma unroll
        for (int c = 0; c < 16; ++c) { ur[c] = u1s[t * 16 + c]; zden += ur[c] * cs[c]; }
        float z = 1.0f / zden, rs = 0.f;
        #pragma unroll
        for (int d = 0; d < 16; ++d) {
            float s = 0.f;
            #pragma unroll
            for (int c = 0; c < 16; ++c) s += ur[c] * kv[c * 16 + d];
            s *= z; xb[t * 16 + d] = s; rs += s;
        }
        rx[t] = rs;
    }
    __syncthreads();

    if (t < NROW) {                       // conv collapsed to Weff dot + row sum
        const int g = t >> 4;
        const float* wrow = weff + t * 256;
        const float* xg = xb + g * 256;
        float tc = 0.f;
        #pragma unroll 4
        for (int e = 0; e < 256; ++e) tc += xg[e] * wrow[e];
        tb[t] = rx[t] + tc + 16.0f * dwc_b[t];
    }
    __syncthreads();

    if (t < NROW) {                       // s[c,q] = sum_p usum[c,p,q] * t[c*16+p]
        int c = t >> 4, q = t & 15; float s = 0.f;
        #pragma unroll
        for (int p = 0; p < 16; ++p) s += usm[c * 256 + p * 16 + q] * tb[c * 16 + p];
        sb[t] = s;
    }
    __syncthreads();

    if (t < NROW) {                       // squash over q
        int c = t >> 4;
        float n2 = 0.f;
        #pragma unroll
        for (int q = 0; q < 16; ++q) { float v = sb[c * 16 + q]; n2 += v * v; }
        float n = sqrtf(n2);
        float coef = 1.0f - 1.0f / (expf(n) + 1e-20f);
        out[(size_t)b * NROW + t] = coef * sb[t] / (n + 1e-20f);
    }
}

// ---------------------------------------------------------------------------
extern "C" void kernel_launch(void* const* d_in, const int* in_sizes, int n_in,
                              void* d_out, int out_size, void* d_ws, size_t ws_size,
                              hipStream_t stream) {
    const float* x1 = (const float*)d_in[0];
    const float* x2 = (const float*)d_in[1];
    const float* x3 = (const float*)d_in[2];
    const float* W1 = (const float*)d_in[3];
    const float* W2 = (const float*)d_in[4];
    const float* W3 = (const float*)d_in[5];
    const float* dwc_w = (const float*)d_in[6];
    const float* dwc_b = (const float*)d_in[7];
    float* out = (float*)d_out;

    // workspace carve-out (~316.5 MB total)
    char* ws = (char*)d_ws;
    size_t off = 0;
    auto take = [&](size_t bytes) -> char* {
        char* p = ws + off;
        off = (off + bytes + 255) & ~(size_t)255;
        return p;
    };
    _Float16* xf    = (_Float16*)take(3ULL * BATCH * NK * 2);
    _Float16* wfp   = (_Float16*)take(3ULL * NCOL  * NK * 2);
    float*    weff  = (float*)   take((size_t)NROW * 256 * 4);
    _Float16* usumg = (_Float16*)take((size_t)BATCH * NCOL * 2);
    _Float16* ufoc  = (_Float16*)take(3ULL * BATCH * NCOL * 2);

    {
        long long total = 3LL * BATCH * NK + 3LL * NCOL * NK;
        int blocks = (int)((total + 255) / 256);
        rc_cvt_kernel<<<blocks, 256, 0, stream>>>(x1, x2, x3, W1, W2, W3, xf, wfp);
    }
    rc_weff_kernel<<<(NROW * 16 + 255) / 256, 256, 0, stream>>>(dwc_w, weff);
    rc_gemm_focus_kernel<<<dim3(BATCH / 16, CLSN), 96, 0, stream>>>(xf, wfp, usumg, ufoc);
    rc_attn_kernel<<<BATCH, 160, 0, stream>>>(ufoc, usumg, weff, dwc_b, out);
}
